// DGCNN_Seg_6390911337308
// MI455X (gfx1250) — matrix operand order repaired and checked by reading, not compiled, not run on hardware
//
#include <hip/hip_runtime.h>
#include <hip/hip_bf16.h>

// ---------------------------------------------------------------------------
// DGCNN-seg forward for MI455X (gfx1250, wave32, WMMA).
// - All GEMMs use v_wmma_f32_16x16x32_f16 (f16 in, f32 accum).
// - Weight tiles staged to LDS with GLOBAL_LOAD_ASYNC_TO_LDS_B128 (ASYNCcnt),
//   A-fragments then come from LDS (ds_load_b128), B-fragments stream from
//   global; each wave computes a 16x64 output block (4 WMMA per A-fragment).
// - KNN keeps a 16x4096 f32 distance tile in LDS (264KB of the 320KB WGP LDS).
// ---------------------------------------------------------------------------

typedef __attribute__((ext_vector_type(16))) _Float16 v16h;
typedef __attribute__((ext_vector_type(8)))  _Float16 v8h;
typedef __attribute__((ext_vector_type(8)))  float    v8f;

#define BB 4
#define NN 4096
#define KNN_K 20
#define EPS 1e-5f
#define NEG_BIG -1e30f

// ---- WMMA fragment loaders (CDNA5 ISA 7.12.2 f16 layouts) ------------------
// A-matrix 16x32 f16: lanes 0-15: M=lane,   K = k0 + {0..7, 16..23}
//                     lanes16-31: M=lane-16,K = k0 + {8..15, 24..31}
__device__ __forceinline__ v16h load_frag_A(const _Float16* __restrict__ base,
                                            int stride, int row, int k0, int lane) {
  const int hs = (lane >> 4) * 8;
  const _Float16* p = base + (size_t)row * stride + k0 + hs;
  v8h lo = *(const v8h*)(p);        // K = k0+hs   .. +7   (16B)
  v8h hi = *(const v8h*)(p + 16);   // K = k0+hs+16.. +23  (16B)
  return __builtin_shufflevector(lo, hi, 0,1,2,3,4,5,6,7,8,9,10,11,12,13,14,15);
}
// B-matrix 32x16 f16: lanes 0-15: N=lane,    K = k0 + {0..7, 16..23}
//                     lanes16-31: N=lane-16, K = k0 + {8..15, 24..31}
__device__ __forceinline__ v16h load_frag_B(const _Float16* __restrict__ base,
                                            int stride, int col, int k0, int lane) {
  const _Float16* p = base + (size_t)col * stride + k0 + ((lane >> 4) << 3);
  v8h lo = *(const v8h*)(p);        // 16B
  v8h hi = *(const v8h*)(p + 16);   // 16B
  return __builtin_shufflevector(lo, hi, 0,1,2,3,4,5,6,7,8,9,10,11,12,13,14,15);
}

__device__ __forceinline__ v8f wmma_f16(v16h a, v16h b, v8f c) {
  return __builtin_amdgcn_wmma_f32_16x16x32_f16(false, a, false, b, (short)0, c,
                                                false, false);
}

// async copy `bytes` (multiple of 16, 16B aligned) global -> LDS, block-wide.
// Uses gfx1250 GLOBAL_LOAD_ASYNC_TO_LDS_B128 (tracked by ASYNCcnt).
__device__ __forceinline__ void async_copy_to_lds(_Float16* lds_dst,
                                                  const _Float16* gsrc,
                                                  int bytes, int tid, int nthreads) {
  unsigned lbase = (unsigned)(uintptr_t)lds_dst;   // low 32 bits = LDS offset
  const char* g = (const char*)gsrc;
  int chunks = bytes >> 4;
  for (int i = tid; i < chunks; i += nthreads) {
    unsigned laddr = lbase + (unsigned)(i << 4);
    unsigned long long gaddr = (unsigned long long)(uintptr_t)(g + ((size_t)i << 4));
    asm volatile("global_load_async_to_lds_b128 %0, %1, off"
                 :: "v"(laddr), "v"(gaddr) : "memory");
  }
  asm volatile("s_wait_asynccnt 0x0" ::: "memory");
}

// ---- generic GEMM: Y[b][o][n] = sum_c W[o][c] * X[b][n][c] -----------------
// block = 128 threads (4 waves). Weight tile (16 rows x C) async-staged in LDS
// and shared by all waves; wave w computes a 16x64 output block (4 n-tiles),
// reusing each A-fragment across 4 WMMAs. C%32==0, O%16==0, N%256==0.
__global__ void k_gemm_f16(const _Float16* __restrict__ W,   // [O][C] f16
                           const _Float16* __restrict__ X,   // [B][N][C] f16
                           float* __restrict__ Y,            // [B][O][N] f32
                           int O, int C, int N) {
  extern __shared__ _Float16 sW[];                 // 16 * C halves
  const int tid  = threadIdx.x;
  const int lane = tid & 31;
  const int wave = tid >> 5;
  const int o0 = blockIdx.y * 16;
  const int b  = blockIdx.z;
  const int n0 = (blockIdx.x * 4 + wave) * 64;     // 64 columns per wave
  const _Float16* Xb = X + (size_t)b * N * C;

  async_copy_to_lds(sW, W + (size_t)o0 * C, 32 * C, tid, 128);
  __syncthreads();

  v8f acc0 = {}, acc1 = {}, acc2 = {}, acc3 = {};
  const int m = lane & 15;
  for (int k0 = 0; k0 < C; k0 += 32) {
    v16h a = load_frag_A(sW, C, m, k0, lane);      // ds_load from staged tile
    __builtin_prefetch(Xb + (size_t)(n0 + m) * C + k0 + 32, 0, 1);
    v16h b0 = load_frag_B(Xb, C, n0 +  0 + m, k0, lane);
    v16h b1 = load_frag_B(Xb, C, n0 + 16 + m, k0, lane);
    v16h b2 = load_frag_B(Xb, C, n0 + 32 + m, k0, lane);
    v16h b3 = load_frag_B(Xb, C, n0 + 48 + m, k0, lane);
    acc0 = wmma_f16(a, b0, acc0);
    acc1 = wmma_f16(a, b1, acc1);
    acc2 = wmma_f16(a, b2, acc2);
    acc3 = wmma_f16(a, b3, acc3);
  }
  // C/D layout: VGPR r -> row o0 + (lane<16 ? r : 8+r), col tile + (lane&15)
  const int mbase = o0 + ((lane >> 4) << 3);
  float* Yb = Y + (size_t)b * O * N;
#pragma unroll
  for (int r = 0; r < 8; ++r) {
    size_t rowoff = (size_t)(mbase + r) * N;
    Yb[rowoff + n0 +  0 + m] = acc0[r];
    Yb[rowoff + n0 + 16 + m] = acc1[r];
    Yb[rowoff + n0 + 32 + m] = acc2[r];
    Yb[rowoff + n0 + 48 + m] = acc3[r];
  }
}

// ---- KNN: per block, 16 query rows; dist tile resident in LDS --------------
// dist[r][j] = 2 * <f_r, f_j> - ||f_j||^2  (row-constant term dropped: same
// ranking as reference's -||fi||^2 + 2<fi,fj> - ||fj||^2). Then 20x argmax.
__global__ void k_knn(const _Float16* __restrict__ F,  // [B][N][64] f16
                      const float* __restrict__ xx,    // [B][N] squared norms
                      int* __restrict__ idx,           // [B][N][20]
                      int N) {
  extern __shared__ char smem[];
  float* dist = (float*)smem;                            // 16*N f32
  float* rv   = (float*)(smem + (size_t)16 * N * 4);     // 256 f32
  int*   ri   = (int*)(rv + 256);                        // 256 i32
  const int lane = threadIdx.x & 31;
  const int wave = threadIdx.x >> 5;                     // 0..7
  const int b  = blockIdx.y;
  const int r0 = blockIdx.x * 16;
  const _Float16* Fb = F + (size_t)b * N * 64;
  const float* xxb = xx + (size_t)b * N;

  const int arow = r0 + (lane & 15);
  v16h a0 = load_frag_A(Fb, 64, arow, 0,  lane);
  v16h a1 = load_frag_A(Fb, 64, arow, 32, lane);
  const int mbase = (lane >> 4) << 3;

  for (int ct = wave; ct < (N >> 4); ct += 8) {
    const int ccol = ct * 16 + (lane & 15);
    v16h b0 = load_frag_B(Fb, 64, ccol, 0,  lane);
    v16h b1 = load_frag_B(Fb, 64, ccol, 32, lane);
    v8f acc = {};
    acc = wmma_f16(a0, b0, acc);
    acc = wmma_f16(a1, b1, acc);
    const float xc = xxb[ccol];
#pragma unroll
    for (int r = 0; r < 8; ++r)
      dist[(size_t)(mbase + r) * N + ccol] = 2.0f * acc[r] - xc;
  }
  __syncthreads();

  // top-20 per row via iterated parallel argmax (lowest index wins ties)
  for (int row = 0; row < 16; ++row) {
    float* drow = dist + (size_t)row * N;
    for (int kk = 0; kk < KNN_K; ++kk) {
      float bv = NEG_BIG; int bi = 0;
      for (int t = threadIdx.x; t < N; t += 256) {
        float v = drow[t];
        if (v > bv) { bv = v; bi = t; }
      }
      rv[threadIdx.x] = bv; ri[threadIdx.x] = bi;
      __syncthreads();
      for (int s = 128; s > 0; s >>= 1) {
        if (threadIdx.x < s) {
          float ov = rv[threadIdx.x + s];
          if (ov > rv[threadIdx.x] ||
              (ov == rv[threadIdx.x] && ri[threadIdx.x + s] < ri[threadIdx.x])) {
            rv[threadIdx.x] = ov; ri[threadIdx.x] = ri[threadIdx.x + s];
          }
        }
        __syncthreads();
      }
      if (threadIdx.x == 0) {
        idx[((size_t)b * N + (r0 + row)) * KNN_K + kk] = ri[0];
        drow[ri[0]] = NEG_BIG;
      }
      __syncthreads();
    }
  }
}

// ---- small elementwise / reduction kernels ---------------------------------
__global__ void k_cast_h(_Float16* dst, const float* src, int n) {
  int i = blockIdx.x * blockDim.x + threadIdx.x;
  if (i < n) dst[i] = (_Float16)src[i];
}

// split edge weight W[o][0:64]=Wa, W[o][64:128]=Wb into Wa and Wd = Wb - Wa
__global__ void k_split_edge_w(_Float16* wa, _Float16* wd, const float* w) {
  int i = blockIdx.x * blockDim.x + threadIdx.x;   // 64*64
  int o = i >> 6, c = i & 63;
  float a = w[o * 128 + c], b = w[o * 128 + 64 + c];
  wa[i] = (_Float16)a;
  wd[i] = (_Float16)(b - a);
}

__global__ void k_conv1(const float* __restrict__ x, const float* __restrict__ W1,
                        float* __restrict__ y, int N) {
  int i = blockIdx.x * blockDim.x + threadIdx.x;   // B*64*N
  int n = i % N, rest = i / N, o = rest & 63, b = rest >> 6;
  const float* xb = x + (size_t)b * 3 * N;
  y[i] = W1[o*3+0]*xb[n] + W1[o*3+1]*xb[N+n] + W1[o*3+2]*xb[2*N+n];
}

// per-channel mean/rstd over [b][C][N]
__global__ void k_stats(const float* __restrict__ y, int C, int N, int B,
                        float* mean, float* rstd) {
  __shared__ float ss[256], sq[256];
  int c = blockIdx.x;
  float s = 0.f, q = 0.f;
  for (int b = 0; b < B; ++b) {
    const float* p = y + ((size_t)b * C + c) * N;
    for (int n = threadIdx.x; n < N; n += 256) { float v = p[n]; s += v; q += v*v; }
  }
  ss[threadIdx.x] = s; sq[threadIdx.x] = q; __syncthreads();
  for (int st = 128; st > 0; st >>= 1) {
    if (threadIdx.x < st) { ss[threadIdx.x] += ss[threadIdx.x+st]; sq[threadIdx.x] += sq[threadIdx.x+st]; }
    __syncthreads();
  }
  if (threadIdx.x == 0) {
    float cnt = (float)B * (float)N;
    float m = ss[0] / cnt, v = sq[0] / cnt - m * m;
    mean[c] = m; rstd[c] = rsqrtf(v + EPS);
  }
}

// BN + LeakyReLU; f32 [b][C][N] -> f16 [b][n][C]
__global__ void k_bnact(const float* __restrict__ y, int C, int N,
                        const float* g, const float* be,
                        const float* mean, const float* rstd,
                        _Float16* __restrict__ out) {
  int i = blockIdx.x * blockDim.x + threadIdx.x;   // B*C*N
  int n = i % N, rest = i / N, c = rest % C, b = rest / C;
  float v = (y[i] - mean[c]) * rstd[c] * g[c] + be[c];
  v = v >= 0.f ? v : 0.2f * v;
  out[((size_t)b * N + n) * C + c] = (_Float16)v;
}

__global__ void k_sqnorm(const _Float16* __restrict__ F, int C, int N,
                         float* __restrict__ xx) {
  int i = blockIdx.x * blockDim.x + threadIdx.x;   // B*N
  const _Float16* p = F + (size_t)i * C;
  float s = 0.f;
  for (int c = 0; c < C; ++c) { float v = (float)p[c]; s += v * v; }
  xx[i] = s;
}

// edge-conv stats: accumulate sum/sumsq of t[o,n,k] = P[o,idx[n,k]] + Q[o,n]
__global__ void k_edge_stats(const float* __restrict__ P, const float* __restrict__ Q,
                             const int* __restrict__ idx, int C, int N,
                             float* ssum, float* ssq) {
  int b = blockIdx.z;
  int n0 = blockIdx.x * 64;
  int c = threadIdx.x;                              // block = 64 threads
  const float* Pb = P + ((size_t)b * C + c) * N;
  const float* Qb = Q + ((size_t)b * C + c) * N;
  const int* ib = idx + (size_t)b * N * KNN_K;
  float s = 0.f, q = 0.f;
  for (int n = n0; n < n0 + 64; ++n) {
    float qv = Qb[n];
    const int* ip = ib + (size_t)n * KNN_K;
    for (int k = 0; k < KNN_K; ++k) { float v = Pb[ip[k]] + qv; s += v; q += v * v; }
  }
  atomicAdd(&ssum[c], s);
  atomicAdd(&ssq[c], q);
}

__global__ void k_finalize(const float* ssum, const float* ssq, int C, float cnt,
                           float* mean, float* rstd) {
  int c = blockIdx.x * blockDim.x + threadIdx.x;
  if (c >= C) return;
  float m = ssum[c] / cnt, v = ssq[c] / cnt - m * m;
  mean[c] = m; rstd[c] = rsqrtf(v + EPS);
}

// edge-conv apply: BN + lrelu + max over k -> f16 [b][n][C]
__global__ void k_edge_apply(const float* __restrict__ P, const float* __restrict__ Q,
                             const int* __restrict__ idx, int C, int N,
                             const float* g, const float* be,
                             const float* mean, const float* rstd,
                             _Float16* __restrict__ out) {
  int i = blockIdx.x * blockDim.x + threadIdx.x;   // B*C*N
  int n = i % N, rest = i / N, c = rest % C, b = rest / C;
  const float* Pb = P + ((size_t)b * C + c) * N;
  float qv = Q[((size_t)b * C + c) * N + n];
  const int* ip = idx + ((size_t)b * N + n) * KNN_K;
  float m = mean[c], rs = rstd[c] * g[c], bb = be[c];
  float best = NEG_BIG;
  for (int k = 0; k < KNN_K; ++k) {
    float v = (Pb[ip[k]] + qv - m) * rs + bb;
    v = v >= 0.f ? v : 0.2f * v;
    best = fmaxf(best, v);
  }
  out[((size_t)b * N + n) * C + c] = (_Float16)best;
}

// layer4: BN + lrelu then max over n -> x6max[b][c]
__global__ void k_bn_maxn(const float* __restrict__ y, int C, int N,
                          const float* g, const float* be,
                          const float* mean, const float* rstd,
                          float* __restrict__ x6max) {
  __shared__ float sm[256];
  int b = blockIdx.y, c = blockIdx.x;
  const float* p = y + ((size_t)b * C + c) * N;
  float m = mean[c], rs = rstd[c] * g[c], bb = be[c];
  float best = NEG_BIG;
  for (int n = threadIdx.x; n < N; n += 256) {
    float v = (p[n] - m) * rs + bb;
    v = v >= 0.f ? v : 0.2f * v;
    best = fmaxf(best, v);
  }
  sm[threadIdx.x] = best; __syncthreads();
  for (int s = 128; s > 0; s >>= 1) {
    if (threadIdx.x < s) sm[threadIdx.x] = fmaxf(sm[threadIdx.x], sm[threadIdx.x + s]);
    __syncthreads();
  }
  if (threadIdx.x == 0) x6max[(size_t)b * C + c] = sm[0];
}

// concat [x1(64) | x2(64) | x3(64) | bcast(x6max,512)] -> f16 [b][n][704]
__global__ void k_concat7(const _Float16* __restrict__ x1h,
                          const _Float16* __restrict__ x2h,
                          const _Float16* __restrict__ x3h,
                          const float* __restrict__ x6max,
                          _Float16* __restrict__ x7h, int N) {
  int i = blockIdx.x * blockDim.x + threadIdx.x;   // B*N*704, c fastest
  int c = i % 704; size_t bn = (size_t)(i / 704);  // = b*N + n
  _Float16 v;
  if      (c < 64)  v = x1h[bn * 64 + c];
  else if (c < 128) v = x2h[bn * 64 + (c - 64)];
  else if (c < 192) v = x3h[bn * 64 + (c - 128)];
  else {
    int b = (int)(bn / N);
    v = (_Float16)x6max[(size_t)b * 512 + (c - 192)];
  }
  x7h[i] = v;
}

// final 2-channel conv + bias
__global__ void k_final8(const float* __restrict__ W8, const float* __restrict__ bias8,
                         const _Float16* __restrict__ xf, float* __restrict__ out, int N) {
  int i = blockIdx.x * blockDim.x + threadIdx.x;   // B*2*N
  int n = i % N, o = (i / N) & 1, b = i / (2 * N);
  const _Float16* f = xf + ((size_t)b * N + n) * 128;
  const float* w = W8 + o * 128;
  float s = 0.f;
  for (int c = 0; c < 128; ++c) s += w[c] * (float)f[c];
  out[i] = s + bias8[o];
}

// ---------------------------------------------------------------------------
extern "C" void kernel_launch(void* const* d_in, const int* in_sizes, int n_in,
                              void* d_out, int out_size, void* d_ws, size_t ws_size,
                              hipStream_t stream) {
  const float* x     = (const float*)d_in[0];
  const float* W1    = (const float*)d_in[1];
  const float* W2    = (const float*)d_in[2];
  const float* W3    = (const float*)d_in[3];
  const float* W4    = (const float*)d_in[4];
  const float* W5    = (const float*)d_in[5];
  const float* W6    = (const float*)d_in[6];
  const float* W7    = (const float*)d_in[7];
  const float* W8    = (const float*)d_in[8];
  const float* bias8 = (const float*)d_in[9];
  const float* g[8]; const float* be[8];
  for (int i = 1; i <= 7; ++i) { g[i] = (const float*)d_in[10 + 2*(i-1)];
                                 be[i] = (const float*)d_in[11 + 2*(i-1)]; }
  float* out = (float*)d_out;
  const int N = NN, B = BB;

  // ---- workspace carve-up (~100 MB) ----
  size_t off = 0;
  auto carve = [&](size_t bytes) { size_t o = off; off = (off + bytes + 255) & ~(size_t)255; return o; };
  char* ws = (char*)d_ws;
  _Float16* x1h  = (_Float16*)(ws + carve((size_t)B*N*64*2));
  _Float16* x2h  = (_Float16*)(ws + carve((size_t)B*N*64*2));
  _Float16* x3h  = (_Float16*)(ws + carve((size_t)B*N*64*2));
  float*    xx   = (float*)   (ws + carve((size_t)B*N*4));
  int*      idxb = (int*)     (ws + carve((size_t)B*N*KNN_K*4));
  float*    P    = (float*)   (ws + carve((size_t)B*64*N*4));
  float*    Q    = (float*)   (ws + carve((size_t)B*64*N*4));
  float*    pre  = (float*)   (ws + carve((size_t)B*512*N*4));   // biggest pre-BN
  _Float16* x7h  = (_Float16*)(ws + carve((size_t)B*N*704*2));
  _Float16* x5h  = (_Float16*)(ws + carve((size_t)B*N*512*2));
  _Float16* x6h2 = (_Float16*)(ws + carve((size_t)B*N*256*2));
  _Float16* xfh  = (_Float16*)(ws + carve((size_t)B*N*128*2));
  float*    mean = (float*)   (ws + carve(512*4));
  float*    rstd = (float*)   (ws + carve(512*4));
  float*    ssum = (float*)   (ws + carve(512*4));
  float*    ssq  = (float*)   (ws + carve(512*4));
  float*    x6mx = (float*)   (ws + carve((size_t)B*512*4));
  _Float16* W2a  = (_Float16*)(ws + carve(64*64*2));
  _Float16* W2d  = (_Float16*)(ws + carve(64*64*2));
  _Float16* W3a  = (_Float16*)(ws + carve(64*64*2));
  _Float16* W3d  = (_Float16*)(ws + carve(64*64*2));
  _Float16* W4h  = (_Float16*)(ws + carve(512*64*2));
  _Float16* W5h  = (_Float16*)(ws + carve((size_t)512*704*2));
  _Float16* W6h  = (_Float16*)(ws + carve(256*512*2));
  _Float16* W7h  = (_Float16*)(ws + carve(128*256*2));
  (void)ws_size; (void)n_in; (void)in_sizes; (void)out_size;

  const size_t knn_shmem = (size_t)16 * N * 4 + 256 * 4 + 256 * 4;  // 264 KB < 320 KB WGP LDS
  (void)hipFuncSetAttribute((const void*)k_knn,
                            hipFuncAttributeMaxDynamicSharedMemorySize, (int)knn_shmem);

  // ---- weight prep ----
  k_cast_h<<<(512*64+255)/256, 256, 0, stream>>>(W4h, W4, 512*64);
  k_cast_h<<<(512*704+255)/256, 256, 0, stream>>>(W5h, W5, 512*704);
  k_cast_h<<<(256*512+255)/256, 256, 0, stream>>>(W6h, W6, 256*512);
  k_cast_h<<<(128*256+255)/256, 256, 0, stream>>>(W7h, W7, 128*256);
  k_split_edge_w<<<(64*64)/256, 256, 0, stream>>>(W2a, W2d, W2);
  k_split_edge_w<<<(64*64)/256, 256, 0, stream>>>(W3a, W3d, W3);

  // ---- layer 1: 3->64 conv + BN + lrelu ----
  k_conv1<<<(B*64*N)/256, 256, 0, stream>>>(x, W1, pre, N);
  k_stats<<<64, 256, 0, stream>>>(pre, 64, N, B, mean, rstd);
  k_bnact<<<(B*64*N)/256, 256, 0, stream>>>(pre, 64, N, g[1], be[1], mean, rstd, x1h);

  // ---- edge conv 2 ----
  k_sqnorm<<<(B*N)/256, 256, 0, stream>>>(x1h, 64, N, xx);
  k_knn<<<dim3(N/16, B), 256, knn_shmem, stream>>>(x1h, xx, idxb, N);
  k_gemm_f16<<<dim3(N/256, 4, B), 128, 16*64*2, stream>>>(W2a, x1h, P, 64, 64, N);
  k_gemm_f16<<<dim3(N/256, 4, B), 128, 16*64*2, stream>>>(W2d, x1h, Q, 64, 64, N);
  hipMemsetAsync(ssum, 0, 512*4, stream);
  hipMemsetAsync(ssq,  0, 512*4, stream);
  k_edge_stats<<<dim3(N/64, 1, B), 64, 0, stream>>>(P, Q, idxb, 64, N, ssum, ssq);
  k_finalize<<<1, 64, 0, stream>>>(ssum, ssq, 64, (float)B*N*KNN_K, mean, rstd);
  k_edge_apply<<<(B*64*N)/256, 256, 0, stream>>>(P, Q, idxb, 64, N, g[2], be[2], mean, rstd, x2h);

  // ---- edge conv 3 ----
  k_sqnorm<<<(B*N)/256, 256, 0, stream>>>(x2h, 64, N, xx);
  k_knn<<<dim3(N/16, B), 256, knn_shmem, stream>>>(x2h, xx, idxb, N);
  k_gemm_f16<<<dim3(N/256, 4, B), 128, 16*64*2, stream>>>(W3a, x2h, P, 64, 64, N);
  k_gemm_f16<<<dim3(N/256, 4, B), 128, 16*64*2, stream>>>(W3d, x2h, Q, 64, 64, N);
  hipMemsetAsync(ssum, 0, 512*4, stream);
  hipMemsetAsync(ssq,  0, 512*4, stream);
  k_edge_stats<<<dim3(N/64, 1, B), 64, 0, stream>>>(P, Q, idxb, 64, N, ssum, ssq);
  k_finalize<<<1, 64, 0, stream>>>(ssum, ssq, 64, (float)B*N*KNN_K, mean, rstd);
  k_edge_apply<<<(B*64*N)/256, 256, 0, stream>>>(P, Q, idxb, 64, N, g[3], be[3], mean, rstd, x3h);

  // ---- layer 4: 64->512, BN + lrelu + global max over n ----
  k_gemm_f16<<<dim3(N/256, 32, B), 128, 16*64*2, stream>>>(W4h, x3h, pre, 512, 64, N);
  k_stats<<<512, 256, 0, stream>>>(pre, 512, N, B, mean, rstd);
  k_bn_maxn<<<dim3(512, B), 256, 0, stream>>>(pre, 512, N, g[4], be[4], mean, rstd, x6mx);

  // ---- concat -> 704ch ----
  k_concat7<<<(B*N*704)/256, 256, 0, stream>>>(x1h, x2h, x3h, x6mx, x7h, N);

  // ---- layer 5: 704->512 ----
  k_gemm_f16<<<dim3(N/256, 32, B), 128, 16*704*2, stream>>>(W5h, x7h, pre, 512, 704, N);
  k_stats<<<512, 256, 0, stream>>>(pre, 512, N, B, mean, rstd);
  k_bnact<<<(B*512*N)/256, 256, 0, stream>>>(pre, 512, N, g[5], be[5], mean, rstd, x5h);

  // ---- layer 6: 512->256 ----
  k_gemm_f16<<<dim3(N/256, 16, B), 128, 16*512*2, stream>>>(W6h, x5h, pre, 256, 512, N);
  k_stats<<<256, 256, 0, stream>>>(pre, 256, N, B, mean, rstd);
  k_bnact<<<(B*256*N)/256, 256, 0, stream>>>(pre, 256, N, g[6], be[6], mean, rstd, x6h2);

  // ---- layer 7: 256->128 ----
  k_gemm_f16<<<dim3(N/256, 8, B), 128, 16*256*2, stream>>>(W7h, x6h2, pre, 128, 256, N);
  k_stats<<<128, 256, 0, stream>>>(pre, 128, N, B, mean, rstd);
  k_bnact<<<(B*128*N)/256, 256, 0, stream>>>(pre, 128, N, g[7], be[7], mean, rstd, xfh);

  // ---- output: 128->2 + bias ----
  k_final8<<<(B*2*N)/256, 256, 0, stream>>>(W8, bias8, xfh, out, N);
}
